// ALIGNN_67585605370457
// MI455X (gfx1250) — matrix operand order repaired and checked by reading, not compiled, not run on hardware
//
#include <hip/hip_runtime.h>
#include <hip/hip_bf16.h>

// ALIGNN forward for MI455X (gfx1250, wave32, WMMA).
// Dense matmuls: v_wmma_f32_16x16x32_f16 (f16 in, f32 acc), weight panels
// staged in LDS in a fragment-swizzled layout so each B fragment is one
// contiguous 32-byte LDS read (2x ds_load_b128) per K-chunk.
// Scatter/segment sums: global f32 atomics (HBM-bound, 23.3 TB/s ceiling).
//
// Workspace requirement: 64*(50000 + 500000 + 800000 + 4*500000 + 800000
//                             + 2*500000) floats  ~= 1.32 GB.

typedef __attribute__((ext_vector_type(16))) _Float16 v16h;
typedef __attribute__((ext_vector_type(8)))  float    v8f;

// Swizzled LDS index for a weight element (k, n):
//   nt = n>>4 (col tile), col = n&15, kc = k>>5 (K chunk of 32),
//   hh = (k>>3)&1 (half-wave), j = (k&7) | (((k>>4)&1)<<3).
// halves for one lane's fragment (fixed nt,kc,hh,col; j=0..15) are contiguous.
template <int KCH>
__device__ __forceinline__ int swz(int k, int n) {
  int nt = n >> 4, col = n & 15;
  int kc = k >> 5, hh = (k >> 3) & 1, j = (k & 7) | (((k >> 4) & 1) << 3);
  return ((((nt * KCH) + kc) * 2 + hh) * 16 + col) * 16 + j;
}

// ---------------------------------------------------------------------------
// Generic WMMA GEMM:  Y[M,NOUT] = X[M,64] @ W[64,NOUT] + bias
// Block = 256 threads = 8 waves; wave w owns row tile (blockIdx*8+w)*16.
// M must be a multiple of 16 (true for all call sites here).
// ---------------------------------------------------------------------------
template <int NOUT>
__launch_bounds__(256)
__global__ void gemm_kernel(const float* __restrict__ X,
                            const float* __restrict__ W,
                            const float* __restrict__ bias,
                            float* __restrict__ Y, int M) {
  constexpr int KCH = 2;                  // K = 64 = 2 x 32
  __shared__ _Float16 sW[64 * NOUT];      // swizzled
  for (int t = threadIdx.x; t < 64 * NOUT; t += blockDim.x) {
    int k = t / NOUT, n = t % NOUT;       // NOUT is a power of two
    sW[swz<KCH>(k, n)] = (_Float16)W[t];
  }
  __syncthreads();

  const int wave = threadIdx.x >> 5;
  const int lane = threadIdx.x & 31;
  const int row0 = (blockIdx.x * 8 + wave) * 16;
  if (row0 >= M) return;

  const int lh = lane & 15;   // row / col within tile
  const int hh = lane >> 4;   // half-wave select

  // A fragment: 16x32 f16; half j -> k = hh*8 + (j<8 ? j : j+8).
  // j=0..7 and j=8..15 are each 8 contiguous floats -> float4 pairs.
  const float4* xv = (const float4*)(X + (size_t)(row0 + lh) * 64);
  float4 p0 = xv[hh * 2 + 0], p1 = xv[hh * 2 + 1];   // k = hh*8 + 0..7
  float4 p2 = xv[hh * 2 + 4], p3 = xv[hh * 2 + 5];   // k = hh*8 + 16..23
  float4 q0 = xv[hh * 2 + 8], q1 = xv[hh * 2 + 9];   // +32
  float4 q2 = xv[hh * 2 + 12], q3 = xv[hh * 2 + 13];
  v16h a0, a1;
  a0[0] = (_Float16)p0.x;  a0[1] = (_Float16)p0.y;
  a0[2] = (_Float16)p0.z;  a0[3] = (_Float16)p0.w;
  a0[4] = (_Float16)p1.x;  a0[5] = (_Float16)p1.y;
  a0[6] = (_Float16)p1.z;  a0[7] = (_Float16)p1.w;
  a0[8] = (_Float16)p2.x;  a0[9] = (_Float16)p2.y;
  a0[10] = (_Float16)p2.z; a0[11] = (_Float16)p2.w;
  a0[12] = (_Float16)p3.x; a0[13] = (_Float16)p3.y;
  a0[14] = (_Float16)p3.z; a0[15] = (_Float16)p3.w;
  a1[0] = (_Float16)q0.x;  a1[1] = (_Float16)q0.y;
  a1[2] = (_Float16)q0.z;  a1[3] = (_Float16)q0.w;
  a1[4] = (_Float16)q1.x;  a1[5] = (_Float16)q1.y;
  a1[6] = (_Float16)q1.z;  a1[7] = (_Float16)q1.w;
  a1[8] = (_Float16)q2.x;  a1[9] = (_Float16)q2.y;
  a1[10] = (_Float16)q2.z; a1[11] = (_Float16)q2.w;
  a1[12] = (_Float16)q3.x; a1[13] = (_Float16)q3.y;
  a1[14] = (_Float16)q3.z; a1[15] = (_Float16)q3.w;

#pragma unroll
  for (int nt = 0; nt < NOUT / 16; ++nt) {
    const int col = nt * 16 + lh;
    // B fragment: one contiguous 32-byte LDS read per K-chunk.
    const _Float16* bbase = &sW[(((nt * KCH) * 2 + hh) * 16 + lh) * 16];
    v16h b0 = *(const v16h*)(bbase);          // kc = 0
    v16h b1 = *(const v16h*)(bbase + 512);    // kc = 1 (2*16*16 halves)
    v8f c = {};
    c = __builtin_amdgcn_wmma_f32_16x16x32_f16(false, a0, false, b0,
                                               (short)0, c, false, false);
    c = __builtin_amdgcn_wmma_f32_16x16x32_f16(false, a1, false, b1,
                                               (short)0, c, false, false);
    const float bb = bias[col];
    // C/D layout: VGPR r -> row = row0 + hh*8 + r, col = nt*16 + lh
#pragma unroll
    for (int r = 0; r < 8; ++r)
      Y[(size_t)(row0 + hh * 8 + r) * NOUT + col] = c[r] + bb;
  }
}

// ---------------------------------------------------------------------------
// Fused RBF + GEMM:  Y[M,64] = rbf(x; vmin,vmax,BINS) @ W[BINS,64] + bias
// A-fragment elements synthesized on the fly: exp(-gamma*(x - c_k)^2),
// K padded with zeros to KCH*32. No [M,BINS] materialization.
// ---------------------------------------------------------------------------
template <int BINS, int KCH>
__launch_bounds__(256)
__global__ void rbf_gemm_kernel(const float* __restrict__ x,
                                const float* __restrict__ W,
                                const float* __restrict__ bias,
                                float* __restrict__ Y, int M,
                                float vmin, float vmax) {
  constexpr int KTOT = KCH * 32;
  __shared__ _Float16 sW[KTOT * 64];        // swizzled
  for (int t = threadIdx.x; t < KTOT * 64; t += blockDim.x) {
    int k = t >> 6, n = t & 63;
    sW[swz<KCH>(k, n)] =
        (k < BINS) ? (_Float16)W[k * 64 + n] : (_Float16)0.f;
  }
  __syncthreads();

  const int wave = threadIdx.x >> 5;
  const int lane = threadIdx.x & 31;
  const int row0 = (blockIdx.x * 8 + wave) * 16;
  if (row0 >= M) return;

  const int lh = lane & 15;
  const int hh = lane >> 4;

  const float d     = x[row0 + lh];
  const float step  = (vmax - vmin) / (float)(BINS - 1);
  const float gamma = (float)(BINS - 1) / (vmax - vmin);

  v16h a[KCH];
#pragma unroll
  for (int ci = 0; ci < KCH; ++ci) {
#pragma unroll
    for (int j = 0; j < 16; ++j) {
      int k = ci * 32 + hh * 8 + (j < 8 ? j : j + 8);
      float v = 0.f;
      if (k < BINS) {
        float t = d - (vmin + (float)k * step);
        v = __expf(-gamma * t * t);
      }
      a[ci][j] = (_Float16)v;
    }
  }

#pragma unroll
  for (int nt = 0; nt < 4; ++nt) {
    const int col = nt * 16 + lh;
    const _Float16* bbase = &sW[(((nt * KCH) * 2 + hh) * 16 + lh) * 16];
    v8f c = {};
#pragma unroll
    for (int ci = 0; ci < KCH; ++ci) {
      v16h b = *(const v16h*)(bbase + ci * 512);
      c = __builtin_amdgcn_wmma_f32_16x16x32_f16(false, a[ci], false, b,
                                                 (short)0, c, false, false);
    }
    const float bb = bias[col];
#pragma unroll
    for (int r = 0; r < 8; ++r)
      Y[(size_t)(row0 + hh * 8 + r) * 64 + col] = c[r] + bb;
  }
}

// ---------------------------------------------------------------------------
// Elementwise / scatter kernels (HBM-bound)
// ---------------------------------------------------------------------------
__global__ void embed_kernel(const int* __restrict__ an,
                             const float* __restrict__ emb,
                             float* __restrict__ h, long total) {
  long i = (long)blockIdx.x * blockDim.x + threadIdx.x;
  if (i >= total) return;
  int node = (int)(i >> 6), f = (int)(i & 63);
  h[i] = emb[an[node] * 64 + f];
}

// e_new = Dh[src] + Eh[dst] + Ce;  sigma = sigmoid(e_new)
// sum_sh[dst] += Bh[src]*sigma;  sum_s[dst] += sigma;
// if write_e: e += silu(e_new) = e_new * sigma  (in place)
__global__ void edge_update_kernel(const int* __restrict__ src,
                                   const int* __restrict__ dst,
                                   const float* __restrict__ Bh,
                                   const float* __restrict__ Dh,
                                   const float* __restrict__ Eh,
                                   const float* __restrict__ Ce,
                                   float* __restrict__ e,
                                   float* __restrict__ sum_sh,
                                   float* __restrict__ sum_s,
                                   long total, int write_e) {
  long i = (long)blockIdx.x * blockDim.x + threadIdx.x;
  if (i >= total) return;
  int eidx = (int)(i >> 6), f = (int)(i & 63);
  int s = src[eidx], d = dst[eidx];
  float en  = Dh[(size_t)s * 64 + f] + Eh[(size_t)d * 64 + f] + Ce[i];
  float sig = 1.f / (1.f + __expf(-en));
  atomicAdd(&sum_sh[(size_t)d * 64 + f], Bh[(size_t)s * 64 + f] * sig);
  atomicAdd(&sum_s[(size_t)d * 64 + f], sig);
  if (write_e) e[i] = e[i] + en * sig;
}

// h += silu(Ah + sum_sh/(sum_s + 1e-6))
__global__ void node_update_kernel(float* __restrict__ h,
                                   const float* __restrict__ Ah,
                                   const float* __restrict__ sum_sh,
                                   const float* __restrict__ sum_s,
                                   long total) {
  long i = (long)blockIdx.x * blockDim.x + threadIdx.x;
  if (i >= total) return;
  float v = Ah[i] + sum_sh[i] / (sum_s[i] + 1e-6f);
  h[i] = h[i] + v * (1.f / (1.f + __expf(-v)));
}

// out: [B,L,256] feats then [B,L] mask. One block per (b,p) row.
__global__ void scatter_out_kernel(const float* __restrict__ node_out,
                                   const int* __restrict__ lens,
                                   float* __restrict__ out,
                                   int nB, int L, int F, int N) {
  int bp = blockIdx.x;
  int b = bp / L, p = bp - b * L;
  int off = 0;
  for (int i = 0; i < b; ++i) off += lens[i];
  const bool valid = p < lens[b];
  int node = off + p;
  node = node < N - 1 ? node : N - 1;
  const float* srcp = node_out + (size_t)node * F;
  float* dstp = out + (size_t)bp * F;
  for (int f = threadIdx.x; f < F; f += blockDim.x)
    dstp[f] = valid ? srcp[f] : 0.f;
  if (threadIdx.x == 0)
    out[(size_t)nB * L * F + bp] = valid ? 1.f : 0.f;
}

// ---------------------------------------------------------------------------
// Host orchestration
// ---------------------------------------------------------------------------
struct GcnBufs { float *A, *B, *D, *E, *C, *ssh, *ss; };

static inline void gemm64(hipStream_t st, const float* X, const float* W,
                          const float* b, float* Y, int M) {
  gemm_kernel<64><<<(M + 127) / 128, 256, 0, st>>>(X, W, b, Y, M);
}

static void gated_gcn(hipStream_t st, const int* src, const int* dst,
                      int n, int ne, float* h, float* e,
                      const float* W5, const float* b5,
                      const GcnBufs& w, bool write_e) {
  gemm64(st, h, W5 + 0 * 4096, b5 + 0 * 64, w.A, n);
  gemm64(st, h, W5 + 1 * 4096, b5 + 1 * 64, w.B, n);
  gemm64(st, e, W5 + 2 * 4096, b5 + 2 * 64, w.C, ne);
  gemm64(st, h, W5 + 3 * 4096, b5 + 3 * 64, w.D, n);
  gemm64(st, h, W5 + 4 * 4096, b5 + 4 * 64, w.E, n);
  hipMemsetAsync(w.ssh, 0, (size_t)n * 64 * sizeof(float), st);
  hipMemsetAsync(w.ss, 0, (size_t)n * 64 * sizeof(float), st);
  long te = (long)ne * 64;
  edge_update_kernel<<<(unsigned)((te + 255) / 256), 256, 0, st>>>(
      src, dst, w.B, w.D, w.E, w.C, e, w.ssh, w.ss, te, write_e ? 1 : 0);
  long tn = (long)n * 64;
  node_update_kernel<<<(unsigned)((tn + 255) / 256), 256, 0, st>>>(
      h, w.A, w.ssh, w.ss, tn);
}

extern "C" void kernel_launch(void* const* d_in, const int* in_sizes, int n_in,
                              void* d_out, int out_size, void* d_ws,
                              size_t ws_size, hipStream_t stream) {
  const int*   atomic_number = (const int*)d_in[0];
  const int*   src           = (const int*)d_in[1];
  const int*   dst           = (const int*)d_in[2];
  const int*   lg_src        = (const int*)d_in[3];
  const int*   lg_dst        = (const int*)d_in[4];
  const float* distance      = (const float*)d_in[5];
  const float* angle         = (const float*)d_in[6];
  const int*   crystal_len   = (const int*)d_in[7];
  const float* node_emb      = (const float*)d_in[8];
  const float* edge_W        = (const float*)d_in[9];
  const float* edge_b        = (const float*)d_in[10];
  const float* ang_W         = (const float*)d_in[11];
  const float* ang_b         = (const float*)d_in[12];
  const float* Wg            = (const float*)d_in[13];  // [6,5,64,64]
  const float* bg            = (const float*)d_in[14];  // [6,5,64]
  const float* fc_W          = (const float*)d_in[15];  // [64,256]
  const float* fc_b          = (const float*)d_in[16];  // [256]

  const int nN = in_sizes[0];   // 50000
  const int nE = in_sizes[1];   // 500000
  const int nL = in_sizes[3];   // 800000
  const int nB = in_sizes[7];   // 50
  const int L = 1024, F = 256;

  // Workspace carve-up (floats). A..E sized for line-graph n (=nE),
  // C sized for line-graph edges (=nL).
  float* ws = (float*)d_ws;
  size_t o = 0;
  float* h  = ws + o; o += (size_t)nN * 64;
  float* e  = ws + o; o += (size_t)nE * 64;
  float* l  = ws + o; o += (size_t)nL * 64;
  GcnBufs wb;
  wb.A   = ws + o; o += (size_t)nE * 64;
  wb.B   = ws + o; o += (size_t)nE * 64;
  wb.D   = ws + o; o += (size_t)nE * 64;
  wb.E   = ws + o; o += (size_t)nE * 64;
  wb.C   = ws + o; o += (size_t)nL * 64;
  wb.ssh = ws + o; o += (size_t)nE * 64;
  wb.ss  = ws + o; o += (size_t)nE * 64;

  // 1) node embedding gather
  {
    long t = (long)nN * 64;
    embed_kernel<<<(unsigned)((t + 255) / 256), 256, 0, stream>>>(
        atomic_number, node_emb, h, t);
  }

  // 2) RBF projections (fused RBF + WMMA GEMM)
  rbf_gemm_kernel<80, 3><<<(nE + 127) / 128, 256, 0, stream>>>(
      distance, edge_W, edge_b, e, nE, 0.f, 8.f);
  rbf_gemm_kernel<40, 2><<<(nL + 127) / 128, 256, 0, stream>>>(
      angle, ang_W, ang_b, l, nL, -1.f, 1.f);

  // 3) ALIGNN layers: graph gcn (h,e) then line-graph gcn (e,l)
  for (int i = 0; i < 2; ++i) {
    gated_gcn(stream, src, dst, nN, nE, h, e,
              Wg + (size_t)(2 * i) * 5 * 4096, bg + (size_t)(2 * i) * 5 * 64,
              wb, true);
    gated_gcn(stream, lg_src, lg_dst, nE, nL, e, l,
              Wg + (size_t)(2 * i + 1) * 5 * 4096,
              bg + (size_t)(2 * i + 1) * 5 * 64, wb, true);
  }
  // trailing GatedGCN layers: update h only (reference discards e_out)
  for (int i = 0; i < 2; ++i) {
    gated_gcn(stream, src, dst, nN, nE, h, e,
              Wg + (size_t)(4 + i) * 5 * 4096, bg + (size_t)(4 + i) * 5 * 64,
              wb, false);
  }

  // 4) FC: node_out[nN,256] = h @ fc_W + fc_b  (reuse wb.A, layers done)
  float* node_out = wb.A;
  gemm_kernel<256><<<(nN + 127) / 128, 256, 0, stream>>>(
      h, fc_W, fc_b, node_out, nN);

  // 5) batched scatter + mask
  scatter_out_kernel<<<nB * L, 256, 0, stream>>>(
      node_out, crystal_len, (float*)d_out, nB, L, F, nN);
}
